// Routing_12515534700777
// MI455X (gfx1250) — matrix operand order repaired and checked by reading, not compiled, not run on hardware
//
#include <hip/hip_runtime.h>

typedef __attribute__((ext_vector_type(16))) __bf16  v16bf;
typedef __attribute__((ext_vector_type(8)))  __bf16  v8bf;
typedef __attribute__((ext_vector_type(2)))  __bf16  v2bf;
typedef __attribute__((ext_vector_type(8)))  float   v8f;
typedef __attribute__((ext_vector_type(4)))  float   f4;

#define NB    32          // batch
#define NIN   2048        // input capsules
#define NK    32          // in_dim (WMMA K)
#define NOUT  64          // output capsules
#define ND    32          // out_dim

#define WO_STR  2097152   // weight o-stride (NIN*ND*NK)
#define WI_STR  1024      // weight i-stride (ND*NK)
#define XB_STR  65536     // x b-stride (NIN*NK)

#define VS_STRIDE 2052    // padded vsum row per b (64*32 + 4) -> bank-conflict free
#define LG_STRIDE 68      // padded logits row per b (64 + 4)
#define XA_STRIDE 40      // padded f32 x-stage row (32 + 8) : 160B, 16B aligned
#define XB_STRIDE 40      // padded bf16 x row (32 + 8) : 80B, 16B aligned

// LDS float offsets
#define VS_OFF 0
#define LG_OFF (NB*VS_STRIDE)                  // 65664
#define XA_OFF (LG_OFF + NB*LG_STRIDE)         // 67840
#define LDS_FLOATS (XA_OFF + NB*XA_STRIDE)     // 69120 floats
#define XBF_OFF_BYTES (LDS_FLOATS * 4)         // 276480 (16B aligned)
#define LDS_BYTES (XBF_OFF_BYTES + NB*XB_STRIDE*2)  // 279040 B < 320KB

// Async DMA of one x[:, i, :] slice (4KB) into the f32 LDS staging area.
// Issued early so it overlaps compute; completion via ASYNCcnt.
__device__ __forceinline__ void issue_x_async(const float* __restrict__ x,
                                              float* xa, int i, int tid) {
  if (tid < 256) {
    int b = tid >> 3, seg = tid & 7;                 // 8 x b128 per 128B row
    unsigned ldsaddr = (unsigned)(size_t)(&xa[b * XA_STRIDE + seg * 4]);
    unsigned long long ga =
        (unsigned long long)(size_t)(x + (size_t)b * XB_STR + (size_t)i * NK + seg * 4);
    asm volatile("global_load_async_to_lds_b128 %0, %1, off"
                 :: "v"(ldsaddr), "v"(ga) : "memory");
  }
}
__device__ __forceinline__ void wait_async0() {
  asm volatile("s_wait_asynccnt 0x0" ::: "memory");
}

// One routing pass: recompute u_hat tiles with bf16 WMMA, logits = u.vsum,
// softmax over o, accumulate s = sum_i c*u_hat into per-workgroup partials.
__global__ __launch_bounds__(1024, 1) void caps_pass_kernel(
    const float* __restrict__ x, const float* __restrict__ w,
    const float* __restrict__ vsum, float* __restrict__ spart)
{
  extern __shared__ float smem[];
  float*  vs  = smem + VS_OFF;
  float*  lg  = smem + LG_OFF;
  float*  xa  = smem + XA_OFF;                         // f32 staging (DMA target)
  __bf16* xbf = (__bf16*)((char*)smem + XBF_OFF_BYTES);// packed bf16 x

  const int tid  = threadIdx.x;
  const int lane = tid & 31;
  const int wave = tid >> 5;     // 0..31, owns o = 2*wave .. 2*wave+1
  const int l15  = lane & 15;
  const int kh   = lane >> 4;    // lane half selects K sub-range (ISA A/B layout)

  // stage vsum -> LDS (padded rows per b)
  for (int t = tid; t < NB*NOUT*ND; t += 1024) {
    int b    = t >> 11;          // /2048
    int rest = t & 2047;         // o*32+d
    vs[b*VS_STRIDE + rest] = vsum[t];
  }

  // loop-invariant LDS addresses (avoid per-iteration spills/recompute)
  const float* vrow[2];          // vsum row base per bt, with kh*8 folded in
  int cidx[2];                   // c / logit row index per bt
  #pragma unroll
  for (int bt = 0; bt < 2; ++bt) {
    vrow[bt] = &vs[(bt*16 + l15)*VS_STRIDE + kh*8];
    cidx[bt] = (bt*16 + l15)*LG_STRIDE;
  }
  const __bf16* xrow[2];
  #pragma unroll
  for (int bt = 0; bt < 2; ++bt) xrow[bt] = xbf + (bt*16 + l15)*XB_STRIDE + kh*8;
  const float* wbase = w + (size_t)(wave*2)*WO_STR;

  v8f sa[2][2][2];               // [oo][dt][bt] s accumulators
  #pragma unroll
  for (int a = 0; a < 2; ++a)
    #pragma unroll
    for (int bq = 0; bq < 2; ++bq)
      #pragma unroll
      for (int cq = 0; cq < 2; ++cq)
        sa[a][bq][cq] = (v8f){0,0,0,0,0,0,0,0};

  const int istep = gridDim.x;
  // prologue: DMA first slice
  if (blockIdx.x < NIN) issue_x_async(x, xa, blockIdx.x, tid);

  for (int i = blockIdx.x; i < NIN; i += istep) {
    wait_async0();
    __syncthreads();             // xa ready everywhere; lg free for reuse

    // pack f32 stage -> bf16 (once per workgroup, not per wave)
    if (tid < 512) {
      int b = tid >> 4, j = (tid & 15) * 2;
      float f0 = xa[b*XA_STRIDE + j], f1 = xa[b*XA_STRIDE + j + 1];
      v2bf pk; pk[0] = (__bf16)f0; pk[1] = (__bf16)f1;
      *(v2bf*)&xbf[b*XB_STRIDE + j] = pk;
    }
    __syncthreads();             // xbf ready; xa free

    // overlap next slice's DMA with this iteration's compute
    if (i + istep < NIN) issue_x_async(x, xa, i + istep, tid);

    // B tiles (K=32 x N=16 batch columns): two aligned 16B LDS loads, no cvt
    v16bf btile[2];
    #pragma unroll
    for (int bt = 0; bt < 2; ++bt) {
      v8bf lo = *(const v8bf*)(xrow[bt]);
      v8bf hi = *(const v8bf*)(xrow[bt] + 16);
      v16bf t;
      #pragma unroll
      for (int j = 0; j < 8; ++j) { t[j] = lo[j]; t[8 + j] = hi[j]; }
      btile[bt] = t;
    }

    v8f   u[2][2][2];            // [oo][dt][bt] u_hat tiles (live across barrier)
    float p[2][2];               // logit partials [oo][bt]

    #pragma unroll
    for (int oo = 0; oo < 2; ++oo) {
      const float* wrow = wbase + (size_t)oo*WO_STR + (size_t)i*WI_STR;
      // A tiles (M=16 rows of d x K=32), ISA 16-bit A layout:
      // lane row m=l15; lanes<16: k in {0..7,16..23}; lanes>=16: k in {8..15,24..31}
      v16bf at[2];
      #pragma unroll
      for (int dt = 0; dt < 2; ++dt) {
        const f4* q = (const f4*)(wrow + (size_t)((dt*16 + l15)*NK + kh*8));
        f4 q0 = q[0], q1 = q[1], q2 = q[4], q3 = q[5];
        v16bf t;
        #pragma unroll
        for (int j = 0; j < 4; ++j) {
          t[j]      = (__bf16)q0[j];
          t[4 + j]  = (__bf16)q1[j];
          t[8 + j]  = (__bf16)q2[j];
          t[12 + j] = (__bf16)q3[j];
        }
        at[dt] = t;
      }
      v8f zero = {0,0,0,0,0,0,0,0};
      #pragma unroll
      for (int dt = 0; dt < 2; ++dt)
        #pragma unroll
        for (int bt = 0; bt < 2; ++bt)
          u[oo][dt][bt] = __builtin_amdgcn_wmma_f32_16x16x32_bf16(
              false, at[dt], false, btile[bt], (short)0, zero, false, false);

      // logits: dot(u_hat[o,:,b], vsum[b,o,:]) ; lane holds d = dt*16+8*kh+r
      const int ocol = (wave*2 + oo) * ND;
      #pragma unroll
      for (int bt = 0; bt < 2; ++bt) {
        float acc = 0.f;
        #pragma unroll
        for (int dt = 0; dt < 2; ++dt) {
          const f4* vq = (const f4*)(vrow[bt] + ocol + dt*16);
          f4 v0 = vq[0], v1 = vq[1];
          #pragma unroll
          for (int r = 0; r < 4; ++r) {
            acc += u[oo][dt][bt][r]     * v0[r];
            acc += u[oo][dt][bt][4 + r] * v1[r];
          }
        }
        acc += __shfl_xor(acc, 16);   // combine the two d-halves (wave32)
        p[oo][bt] = acc;
      }
    }
    if (lane < 16) {
      #pragma unroll
      for (int oo = 0; oo < 2; ++oo)
        #pragma unroll
        for (int bt = 0; bt < 2; ++bt)
          lg[(bt*16 + lane)*LG_STRIDE + wave*2 + oo] = p[oo][bt];
    }
    __syncthreads();

    // softmax over o (64) per b, one lane per b
    if (tid < NB) {
      float* row = &lg[tid*LG_STRIDE];
      float mx = row[0];
      for (int o = 1; o < NOUT; ++o) mx = fmaxf(mx, row[o]);
      float sum = 0.f;
      for (int o = 0; o < NOUT; ++o) { float e = __expf(row[o] - mx); row[o] = e; sum += e; }
      float inv = 1.f / sum;
      for (int o = 0; o < NOUT; ++o) row[o] *= inv;
    }
    __syncthreads();

    // s += c * u_hat
    #pragma unroll
    for (int oo = 0; oo < 2; ++oo) {
      const int o = wave*2 + oo;
      #pragma unroll
      for (int bt = 0; bt < 2; ++bt) {
        const float c = lg[cidx[bt] + o];
        #pragma unroll
        for (int dt = 0; dt < 2; ++dt)
          #pragma unroll
          for (int r = 0; r < 8; ++r)
            sa[oo][dt][bt][r] += c * u[oo][dt][bt][r];
      }
    }
  }

  // write per-workgroup s partial, register-natural layout, fully coalesced
  float* dst = spart + (size_t)blockIdx.x * (NOUT*ND*NB);
  #pragma unroll
  for (int oo = 0; oo < 2; ++oo) {
    const int o = wave*2 + oo;
    #pragma unroll
    for (int dt = 0; dt < 2; ++dt)
      #pragma unroll
      for (int bt = 0; bt < 2; ++bt)
        #pragma unroll
        for (int r = 0; r < 8; ++r)
          dst[o*1024 + dt*512 + bt*256 + r*32 + lane] = sa[oo][dt][bt][r];
  }
}

// reduce partials over workgroups, squash, update vsum, write v to out
__global__ __launch_bounds__(256) void reduce_squash_kernel(
    const float* __restrict__ spart, int nwg,
    float* __restrict__ vsum, float* __restrict__ out)
{
  const int row = blockIdx.x;            // b*64 + o
  const int b = row >> 6, o = row & 63;
  const int tid = threadIdx.x;
  const int d = tid & 31, wsl = tid >> 5;
  // invert the register-natural layout of caps_pass_kernel
  const int dt = d >> 4, m = d & 15, kh2 = m >> 3, r = m & 7;
  const int bt = b >> 4, bi = b & 15;
  const int idx = o*1024 + dt*512 + bt*256 + r*32 + kh2*16 + bi;

  float s = 0.f;
  for (int wg = wsl; wg < nwg; wg += 8)
    s += spart[(size_t)wg*65536 + idx];

  __shared__ float red[8][32];
  red[wsl][d] = s;
  __syncthreads();
  if (tid < 32) {
    float t = 0.f;
    #pragma unroll
    for (int j = 0; j < 8; ++j) t += red[j][d];
    float sq = t * t;
    #pragma unroll
    for (int off = 16; off > 0; off >>= 1) sq += __shfl_xor(sq, off);
    // v = (sq/(1+sq)) * s / sqrt(sq)
    float scale = (sq * rsqrtf(sq)) / (1.0f + sq);
    float v = t * scale;
    out[row*ND + d]   = v;
    vsum[row*ND + d] += v;
  }
}

__global__ void zero_kernel(float* p, int n) {
  int t = blockIdx.x * blockDim.x + threadIdx.x;
  if (t < n) p[t] = 0.f;
}

extern "C" void kernel_launch(void* const* d_in, const int* in_sizes, int n_in,
                              void* d_out, int out_size, void* d_ws, size_t ws_size,
                              hipStream_t stream) {
  const float* x = (const float*)d_in[0];   // [32, 2048, 32] f32
  const float* w = (const float*)d_in[1];   // [64, 2048, 32, 32] f32
  float* vsum  = (float*)d_ws;              // 65536 f32
  float* spart = vsum + 65536;              // nwg * 65536 f32

  long slots = ((long)(ws_size / 4) - 65536) / 65536;
  int nwg = (int)(slots < 1 ? 1 : (slots > 256 ? 256 : slots));

  zero_kernel<<<64, 1024, 0, stream>>>(vsum, 65536);

  for (int it = 0; it < 3; ++it) {
    caps_pass_kernel<<<nwg, 1024, (size_t)LDS_BYTES, stream>>>(x, w, vsum, spart);
    reduce_squash_kernel<<<2048, 256, 0, stream>>>(spart, nwg, vsum, (float*)d_out);
  }
}
